// BayesianFilterMask_32959579029621
// MI455X (gfx1250) — matrix operand order repaired and checked by reading, not compiled, not run on hardware
//
#include <hip/hip_runtime.h>

typedef __attribute__((ext_vector_type(16))) _Float16 v16h;
typedef __attribute__((ext_vector_type(8)))  float    v8f;

__device__ __forceinline__ float eluf(float x)  { return x > 0.f ? x : expm1f(x); }
__device__ __forceinline__ float sigmf(float x) { return 1.f / (1.f + __expf(-x)); }

// Epilogue for one 16x16 accumulator tile.
// D layout: VGPR r -> M = tileM + half*8 + r, N = lane column.
__device__ __forceinline__ void store_tile(float* __restrict__ Db, const v8f& acc,
                                           int tileM, int half, int n,
                                           int M, int N, int ldd,
                                           const float* __restrict__ bias,
                                           int biasMode, int act, int nskip)
{
    if (n < N && n >= nskip) {
#pragma unroll
        for (int r = 0; r < 8; ++r) {
            const int m = tileM + (half << 3) + r;
            if (m < M) {
                float v = acc[r];
                if (biasMode == 1)      v += bias[m];
                else if (biasMode == 2) v += bias[n];
                if (act == 1)      v = eluf(v);
                else if (act == 2) v = tanhf(v);
                Db[m * ldd + (n - nskip)] = v;
            }
        }
    }
}

// ---------------------------------------------------------------------------
// WMMA GEMM: D[m,n] = act( sum_k A(m,k)*B(k,n) + bias ), f16 operands with
// f32 accumulate via v_wmma_f32_16x16x32_f16. One wave computes a 16x32 tile
// of D (two 16x16 accumulators sharing one A fragment).
// Out-of-range M rows / N cols are *clamped* (not branched): they only feed
// discarded D rows/cols, so the K-loop is branch-free. K tail is a single
// branchless select chunk.
// blockIdx.x = batch slice, blockIdx.y = M tile (16), blockIdx.z = N tile (32).
// TRANSA: A(m,k) = A[k*lda + m]   (channel linears with W[cin,cout])
// ---------------------------------------------------------------------------
template<bool TRANSA>
__global__ void __launch_bounds__(32)
gemm_wmma_kernel(const float* __restrict__ A, long long sAz, int lda,
                 const float* __restrict__ Bm, long long sBz, int ldb,
                 float* __restrict__ D, long long sDz, int ldd,
                 int M, int N, int K,
                 const float* __restrict__ bias, int biasMode, int act, int nskip)
{
    const float* Ab = A  + (long long)blockIdx.x * sAz;
    const float* Bb = Bm + (long long)blockIdx.x * sBz;
    float*       Db = D  + (long long)blockIdx.x * sDz;
    const int tileM  = blockIdx.y << 4;
    const int tileN0 = blockIdx.z << 5;
    const int lane = threadIdx.x & 31;
    const int half = lane >> 4;
    const int l16  = lane & 15;

    const int mA  = tileM + l16;
    const int mS  = (mA < M) ? mA : (M - 1);   // clamped: bad rows feed discarded D rows
    const int nB0 = tileN0 + l16;
    const int nB1 = tileN0 + 16 + l16;
    const int nS0 = (nB0 < N) ? nB0 : (N - 1);
    const int nS1 = (nB1 < N) ? nB1 : (N - 1);

    // per-lane base pointers (32-bit offsets thereafter; all extents < 2^31)
    const float* Arow = TRANSA ? (Ab + mS) : (Ab + mS * lda);  // index: k*lda | k
    const float* Bc0  = Bb + nS0;                              // index: k*ldb
    const float* Bc1  = Bb + nS1;

    v8f acc0 = {}, acc1 = {};
    const int kMain = K & ~31;

    for (int k0 = 0; k0 < kMain; k0 += 32) {
        v16h af, bf0, bf1;
#pragma unroll
        for (int j = 0; j < 8; ++j) {   // A 16x32: VGPR j -> K = grp*16 + half*8 + 2*jj
            const int k = k0 + ((j >> 2) << 4) + (half << 3) + ((j & 3) << 1);
            float x0, x1;
            if (TRANSA) { x0 = Arow[k * lda]; x1 = Arow[(k + 1) * lda]; }
            else        { x0 = Arow[k];       x1 = Arow[k + 1]; }
            af[2 * j]     = (_Float16)x0;
            af[2 * j + 1] = (_Float16)x1;
        }
#pragma unroll
        for (int j = 0; j < 8; ++j) {   // B 32x16: VGPR j -> K = grp*16 + half*8 + 2*jj
            const int k = k0 + ((j >> 2) << 4) + (half << 3) + ((j & 3) << 1);
            bf0[2 * j]     = (_Float16)Bc0[k * ldb];
            bf0[2 * j + 1] = (_Float16)Bc0[(k + 1) * ldb];
            bf1[2 * j]     = (_Float16)Bc1[k * ldb];
            bf1[2 * j + 1] = (_Float16)Bc1[(k + 1) * ldb];
        }
        if (k0 + 32 < kMain) {          // uniform-predicated prefetch of next chunk
            __builtin_prefetch(TRANSA ? (const void*)(Arow + (k0 + 32) * lda)
                                      : (const void*)(Arow + k0 + 32), 0, 0);
            __builtin_prefetch((const void*)(Bc0 + (k0 + 32 + (half << 4)) * ldb), 0, 0);
            __builtin_prefetch((const void*)(Bc1 + (k0 + 32 + (half << 4)) * ldb), 0, 0);
        }
        acc0 = __builtin_amdgcn_wmma_f32_16x16x32_f16(false, af, false, bf0,
                                                      (short)0, acc0, false, false);
        acc1 = __builtin_amdgcn_wmma_f32_16x16x32_f16(false, af, false, bf1,
                                                      (short)0, acc1, false, false);
    }

    if (kMain < K) {   // K tail: branchless clamped load + select
        v16h af, bf0, bf1;
#pragma unroll
        for (int j = 0; j < 8; ++j) {
            const int k  = kMain + ((j >> 2) << 4) + (half << 3) + ((j & 3) << 1);
            const int c0 = (k < K) ? k : 0;
            const int c1 = (k + 1 < K) ? (k + 1) : 0;
            float x0 = TRANSA ? Arow[c0 * lda] : Arow[c0];
            float x1 = TRANSA ? Arow[c1 * lda] : Arow[c1];
            x0 = (k < K) ? x0 : 0.f;
            x1 = (k + 1 < K) ? x1 : 0.f;
            af[2 * j]     = (_Float16)x0;
            af[2 * j + 1] = (_Float16)x1;
        }
#pragma unroll
        for (int j = 0; j < 8; ++j) {
            const int k  = kMain + ((j >> 2) << 4) + (half << 3) + ((j & 3) << 1);
            const int c0 = (k < K) ? k : 0;
            const int c1 = (k + 1 < K) ? (k + 1) : 0;
            float y0 = Bc0[c0 * ldb], y1 = Bc0[c1 * ldb];
            float y2 = Bc1[c0 * ldb], y3 = Bc1[c1 * ldb];
            bf0[2 * j]     = (_Float16)((k < K) ? y0 : 0.f);
            bf0[2 * j + 1] = (_Float16)((k + 1 < K) ? y1 : 0.f);
            bf1[2 * j]     = (_Float16)((k < K) ? y2 : 0.f);
            bf1[2 * j + 1] = (_Float16)((k + 1 < K) ? y3 : 0.f);
        }
        acc0 = __builtin_amdgcn_wmma_f32_16x16x32_f16(false, af, false, bf0,
                                                      (short)0, acc0, false, false);
        acc1 = __builtin_amdgcn_wmma_f32_16x16x32_f16(false, af, false, bf1,
                                                      (short)0, acc1, false, false);
    }

    store_tile(Db, acc0, tileM, half, nB0, M, N, ldd, bias, biasMode, act, nskip);
    store_tile(Db, acc1, tileM, half, nB1, M, N, ldd, bias, biasMode, act, nskip);
}

// ---------------------------------------------------------------------------
// Edge-weighted message passing scatter: agg[b,dst,c,t] += x[b,src,c,t]*ea[e]
// ---------------------------------------------------------------------------
__global__ void edge_scatter_kernel(const float* __restrict__ x, const int* __restrict__ ei,
                                    const float* __restrict__ ea, float* __restrict__ agg,
                                    int Bn, int Vn, int Cn, int Tn, int En)
{
    long long idx = (long long)blockIdx.x * blockDim.x + threadIdx.x;
    long long total = (long long)Bn * En * Cn * Tn;
    if (idx >= total) return;
    int t = (int)(idx % Tn); long long r = idx / Tn;
    int c = (int)(r % Cn); r /= Cn;
    int e = (int)(r % En); int b = (int)(r / En);
    int s  = ei[e];
    int dd = ei[En + e];
    float v = x[(((long long)b * Vn + s) * Cn + c) * Tn + t] * ea[e];
    atomicAdd(agg + (((long long)b * Vn + dd) * Cn + c) * Tn + t, v);
}

// xs[t, r, c] = h[r, c, t]
__global__ void to_xs_kernel(const float* __restrict__ h, float* __restrict__ xs,
                             int NV, int C, int T)
{
    long long idx = (long long)blockIdx.x * blockDim.x + threadIdx.x;
    long long total = (long long)T * NV * C;
    if (idx >= total) return;
    int c = (int)(idx % C); long long r2 = idx / C;
    int r = (int)(r2 % NV); int t = (int)(r2 / NV);
    xs[idx] = h[((long long)r * C + c) * T + t];
}

// z[(r*C+c)*T + t] = zs[(t*NV + r)*C + c]
__global__ void zs_to_z_kernel(const float* __restrict__ zs, float* __restrict__ z,
                               int NV, int C, int T)
{
    long long idx = (long long)blockIdx.x * blockDim.x + threadIdx.x;
    long long total = (long long)NV * C * T;
    if (idx >= total) return;
    int t = (int)(idx % T); long long r2 = idx / T;
    int c = (int)(r2 % C); int r = (int)(r2 / C);
    z[idx] = zs[((long long)t * NV + r) * C + c];
}

// GRU gate math: h' = (1-z)*n + z*h
__global__ void gru_gate_kernel(const float* __restrict__ gi, const float* __restrict__ gh,
                                const float* __restrict__ h, float* __restrict__ hn,
                                float* __restrict__ zso, int R, int C)
{
    int idx = blockIdx.x * blockDim.x + threadIdx.x;
    if (idx >= R * C) return;
    int r = idx / C, c = idx % C;
    const float* gr = gi + r * 3 * C;
    const float* hr = gh + r * 3 * C;
    float rg = sigmf(gr[c] + hr[c]);
    float zg = sigmf(gr[C + c] + hr[C + c]);
    float ng = tanhf(gr[2 * C + c] + rg * hr[2 * C + c]);
    float out = (1.f - zg) * ng + zg * h[idx];
    hn[idx] = out;
    if (zso) zso[idx] = out;
}

__global__ void mean_v_kernel(const float* __restrict__ hd, float* __restrict__ hm,
                              int Nn, int Vv, int C)
{
    int idx = blockIdx.x * blockDim.x + threadIdx.x;
    if (idx >= Nn * C) return;
    int n = idx / C, c = idx % C;
    float s = 0.f;
    for (int v = 0; v < Vv; ++v) s += hd[(n * Vv + v) * C + c];
    hm[idx] = s / (float)Vv;
}

__global__ void add_zd_kernel(const float* __restrict__ z, const float* __restrict__ zd,
                              float* __restrict__ o, int Vv, int C, int total)
{
    int idx = blockIdx.x * blockDim.x + threadIdx.x;
    if (idx >= total) return;
    int c = idx % C;
    int n = (idx / C) / Vv;
    o[idx] = z[idx] + zd[n * C + c];
}

__global__ void add2_kernel(const float* __restrict__ a, const float* __restrict__ b,
                            float* __restrict__ o, int total)
{
    int idx = blockIdx.x * blockDim.x + threadIdx.x;
    if (idx >= total) return;
    o[idx] = a[idx] + b[idx];
}

// gin = gm + (t>0 ? henc[..., t-1] : 0); henc flat index (r*C+c)*T + t
__global__ void make_gin_kernel(const float* __restrict__ gm, const float* __restrict__ henc,
                                float* __restrict__ o, int t, int T, int total)
{
    int idx = blockIdx.x * blockDim.x + threadIdx.x;
    if (idx >= total) return;
    float x = (t > 0) ? henc[(long long)idx * T + (t - 1)] : 0.f;
    o[idx] = gm[idx] + x;
}

// out[vc, 0] = 0 ; out[vc, 1+t] = in[vc, t]
__global__ void pad_time_kernel(const float* __restrict__ in, float* __restrict__ o,
                                int T, long long total)
{
    long long idx = (long long)blockIdx.x * blockDim.x + threadIdx.x;
    if (idx >= total) return;
    int T1 = T + 1;
    int tt = (int)(idx % T1);
    long long vc = idx / T1;
    o[idx] = (tt == 0) ? 0.f : in[vc * T + (tt - 1)];
}

// ---------------------------------------------------------------------------
// Host orchestration
// ---------------------------------------------------------------------------
static inline int ceilDiv(int a, int b) { return (a + b - 1) / b; }

static void gemm(hipStream_t st, int nz,
                 const float* A, long long sAz, int lda, int transA,
                 const float* B, long long sBz, int ldb,
                 float* D, long long sDz, int ldd,
                 int M, int N, int K,
                 const float* bias, int biasMode, int act, int nskip)
{
    dim3 g(nz, ceilDiv(M, 16), ceilDiv(N, 32));
    if (transA)
        gemm_wmma_kernel<true><<<g, 32, 0, st>>>(A, sAz, lda, B, sBz, ldb,
                                                 D, sDz, ldd, M, N, K, bias, biasMode, act, nskip);
    else
        gemm_wmma_kernel<false><<<g, 32, 0, st>>>(A, sAz, lda, B, sBz, ldb,
                                                  D, sDz, ldd, M, N, K, bias, biasMode, act, nskip);
}

// channel linear on [2, V, Cin, Tin] -> [2, V, Cout, Tin-nskip]
static void chan(hipStream_t st, const float* in, float* out, int Vn, int Cin, int Cout,
                 const float* W, const float* bias, int act, int Tin, int nskip)
{
    int Tout = Tin - nskip;
    gemm(st, 2 * Vn, W, 0, Cout, 1, in, (long long)Cin * Tin, Tin,
         out, (long long)Cout * Tout, Tout, Cout, Tin, Cin, bias, 1, act, nskip);
}

// pooling: out[b] (Vout x CT) = P (Vout x Vin) @ in[b] (Vin x CT)
static void pool(hipStream_t st, const float* in, float* out, const float* P,
                 int Vout, int Vin, int CT)
{
    gemm(st, 2, P, 0, Vin, 0, in, (long long)Vin * CT, CT,
         out, (long long)Vout * CT, CT, Vout, CT, Vin, nullptr, 0, 0, 0);
}

static void scatter(hipStream_t st, const float* x, const int* ei, const float* ea,
                    float* agg, int Vn, int Cn, int Tn, int En)
{
    long long total = 2LL * En * Cn * Tn;
    int blocks = (int)((total + 255) / 256);
    edge_scatter_kernel<<<blocks, 256, 0, st>>>(x, ei, ea, agg, 2, Vn, Cn, Tn, En);
}

extern "C" void kernel_launch(void* const* d_in, const int* in_sizes, int n_in,
                              void* d_out, int out_size, void* d_ws, size_t ws_size,
                              hipStream_t stream)
{
    (void)in_sizes; (void)out_size;
    if (n_in < 57) return;
    const int T = 150, C = 64, R = 750;          // R = B*V4
    const int V0 = 6000, V1 = 3000, V2 = 1500, V3 = 750, V4 = 375;
    const int E0 = 48000, E1 = 24000, E2 = 12000, E3 = 6000, E4 = 3000;

    const float* x    = (const float*)d_in[0];
    const int*   ei0  = (const int*)d_in[1];  const float* ea0 = (const float*)d_in[2];
    const int*   ei1  = (const int*)d_in[3];  const float* ea1 = (const float*)d_in[4];
    const int*   ei2  = (const int*)d_in[5];  const float* ea2 = (const float*)d_in[6];
    const int*   ei3  = (const int*)d_in[7];  const float* ea3 = (const float*)d_in[8];
    const int*   ei4  = (const int*)d_in[9];  const float* ea4 = (const float*)d_in[10];
    const float* P01 = (const float*)d_in[11], *P12 = (const float*)d_in[12];
    const float* P23 = (const float*)d_in[13], *P34 = (const float*)d_in[14];
    const float* P43 = (const float*)d_in[15], *P32 = (const float*)d_in[16];
    const float* P21 = (const float*)d_in[17], *P10 = (const float*)d_in[18];
    const float* Wc1 = (const float*)d_in[19], *bc1 = (const float*)d_in[20];
    const float* Wc2 = (const float*)d_in[21], *bc2 = (const float*)d_in[22];
    const float* Wc3 = (const float*)d_in[23], *bc3 = (const float*)d_in[24];
    const float* Wc4 = (const float*)d_in[25], *bc4 = (const float*)d_in[26];
    const float* Wfe1 = (const float*)d_in[27], *bfe1 = (const float*)d_in[28];
    const float* Wfe2 = (const float*)d_in[29], *bfe2 = (const float*)d_in[30];
    const float* Wri = (const float*)d_in[31], *Wrh = (const float*)d_in[32];
    const float* bri = (const float*)d_in[33];
    const float* Wd1 = (const float*)d_in[34], *bd1 = (const float*)d_in[35];
    const float* Wd2 = (const float*)d_in[36], *bd2 = (const float*)d_in[37];
    const float* Wp  = (const float*)d_in[38], *bp  = (const float*)d_in[39];
    const float* Wg  = (const float*)d_in[40], *bg  = (const float*)d_in[41];
    const float* Wci = (const float*)d_in[42], *Wch = (const float*)d_in[43];
    const float* bci = (const float*)d_in[44];
    const float* Wfd3 = (const float*)d_in[45], *bfd3 = (const float*)d_in[46];
    const float* Wfd4 = (const float*)d_in[47], *bfd4 = (const float*)d_in[48];
    const float* Wdc4 = (const float*)d_in[49], *bdc4 = (const float*)d_in[50];
    const float* Wdc3 = (const float*)d_in[51], *bdc3 = (const float*)d_in[52];
    const float* Wdc2 = (const float*)d_in[53], *bdc2 = (const float*)d_in[54];
    const float* Wdc1 = (const float*)d_in[55], *bdc1 = (const float*)d_in[56];

    // ---- workspace layout (floats) ----
    const size_t GB = 29200000ULL;  // >= 2*6000*16*151 = 28,992,000
    float* G0 = (float*)d_ws;
    float* G1 = G0 + GB;
    float* G2 = G1 + GB;
    float* HENC = G2 + GB;                 // [2,375,64,150] = 7,200,000
    float* ZS   = HENC + 5760000;          // [150,750,64]   = 7,200,000
    float* SM   = ZS + 7200000;            // small pool (<= 1,000,000)
    if (ws_size < (3 * GB + 5760000 + 7200000 + 1000000) * sizeof(float)) return;

    float* s_h0   = SM;            // 48000
    float* s_h1   = SM + 48000;    // 48000
    float* s_gh   = SM + 96000;    // 144000
    float* s_hd   = SM + 240000;   // 48000
    float* s_hm   = SM + 288000;   // 128
    float* s_zd   = SM + 288256;   // 128
    float* s_t1   = SM + 288512;   // 48000
    float* s_t2   = SM + 336512;   // 48000
    float* s_zode = SM + 384512;   // 48000
    float* s_agg4 = SM + 432512;   // 48000
    float* s_gi   = SM + 480512;   // 144000
    float* s_gh2  = SM + 624512;   // 144000

    // ================= encoder =================
    hipMemsetAsync(G2, 0, (size_t)2 * V0 * 1 * T * 4, stream);
    scatter(stream, x, ei0, ea0, G2, V0, 1, T, E0);
    chan(stream, G2, G1, V0, 1, 16, Wc1, bc1, 1, T, 0);
    pool(stream, G1, G0, P01, V1, V0, 16 * T);

    hipMemsetAsync(G2, 0, (size_t)2 * V1 * 16 * T * 4, stream);
    scatter(stream, G0, ei1, ea1, G2, V1, 16, T, E1);
    chan(stream, G2, G1, V1, 16, 32, Wc2, bc2, 1, T, 0);
    pool(stream, G1, G0, P12, V2, V1, 32 * T);

    hipMemsetAsync(G2, 0, (size_t)2 * V2 * 32 * T * 4, stream);
    scatter(stream, G0, ei2, ea2, G2, V2, 32, T, E2);
    chan(stream, G2, G1, V2, 32, 64, Wc3, bc3, 1, T, 0);
    pool(stream, G1, G0, P23, V3, V2, 64 * T);

    hipMemsetAsync(G2, 0, (size_t)2 * V3 * 64 * T * 4, stream);
    scatter(stream, G0, ei3, ea3, G2, V3, 64, T, E3);
    chan(stream, G2, G1, V3, 64, 128, Wc4, bc4, 1, T, 0);
    pool(stream, G1, G0, P34, V4, V3, 128 * T);

    chan(stream, G0, G1, V4, 128, 256, Wfe1, bfe1, 1, T, 0);
    chan(stream, G1, HENC, V4, 256, 64, Wfe2, bfe2, 2, T, 0);   // tanh

    // ================= RNN domain encoding =================
    {
        long long tot = (long long)T * R * C;
        to_xs_kernel<<<(int)((tot + 255) / 256), 256, 0, stream>>>(HENC, G0, R, C, T);
    }
    // GI[T*R, 192] = xs @ Wri + bri
    gemm(stream, 1, G0, 0, C, 0, Wri, 0, 3 * C, G1, 0, 3 * C,
         T * R, 3 * C, C, bri, 2, 0, 0);
    hipMemsetAsync(s_h0, 0, (size_t)R * C * 4, stream);
    {
        float* cur = s_h0; float* nxt = s_h1;
        int blk = ceilDiv(R * C, 256);
        for (int t = 0; t < T; ++t) {
            gemm(stream, 1, cur, 0, C, 0, Wrh, 0, 3 * C, s_gh, 0, 3 * C,
                 R, 3 * C, C, nullptr, 0, 0, 0);
            gru_gate_kernel<<<blk, 256, 0, stream>>>(G1 + (long long)t * R * 3 * C,
                                                     s_gh, cur, nxt, nullptr, R, C);
            float* tmp = cur; cur = nxt; nxt = tmp;
        }
        // hd = elu(hlast @ Wd1 + bd1); z_D = mean_v(hd) @ Wd2 + bd2
        gemm(stream, 1, cur, 0, C, 0, Wd1, 0, C, s_hd, 0, C, R, C, C, bd1, 2, 1, 0);
        mean_v_kernel<<<1, 128, 0, stream>>>(s_hd, s_hm, 2, V4, C);
        gemm(stream, 1, s_hm, 0, C, 0, Wd2, 0, C, s_zd, 0, C, 2, C, C, bd2, 2, 0, 0);
    }

    // ================= sequential Bayesian filter =================
    {
        float* zc = s_h0; float* zn = s_h1;
        hipMemsetAsync(zc, 0, (size_t)R * C * 4, stream);
        int blk = ceilDiv(R * C, 256);
        for (int t = 0; t < T; ++t) {
            add_zd_kernel<<<blk, 256, 0, stream>>>(zc, s_zd, s_t1, V4, C, R * C);
            gemm(stream, 1, s_t1, 0, C, 0, Wp, 0, C, s_t2, 0, C, R, C, C, bp, 2, 2, 0);
            add2_kernel<<<blk, 256, 0, stream>>>(zc, s_t2, s_zode, R * C);     // Euler step
            hipMemsetAsync(s_agg4, 0, (size_t)R * C * 4, stream);
            scatter(stream, s_zode, ei4, ea4, s_agg4, V4, C, 1, E4);
            gemm(stream, 1, s_agg4, 0, C, 0, Wg, 0, C, s_t1, 0, C, R, C, C, bg, 2, 1, 0);
            make_gin_kernel<<<blk, 256, 0, stream>>>(s_t1, HENC, s_t2, t, T, R * C);
            gemm(stream, 1, s_t2, 0, C, 0, Wci, 0, 3 * C, s_gi, 0, 3 * C,
                 R, 3 * C, C, bci, 2, 0, 0);
            gemm(stream, 1, s_zode, 0, C, 0, Wch, 0, 3 * C, s_gh2, 0, 3 * C,
                 R, 3 * C, C, nullptr, 0, 0, 0);
            gru_gate_kernel<<<blk, 256, 0, stream>>>(s_gi, s_gh2, s_zode, zn,
                                                     ZS + (long long)t * R * C, R, C);
            float* tmp = zc; zc = zn; zn = tmp;
        }
        long long tot = (long long)R * C * T;
        zs_to_z_kernel<<<(int)((tot + 255) / 256), 256, 0, stream>>>(ZS, G0, R, C, T);
    }

    // ================= decoder =================
    chan(stream, G0, G1, V4, 64, 256, Wfd3, bfd3, 1, T, 0);
    chan(stream, G1, G2, V4, 256, 128, Wfd4, bfd4, 1, T, 0);

    // up1: P43, cin=128, ei3, dc4 (128->64)
    pool(stream, G2, G0, P43, V3, V4, 128 * T);
    {
        long long tot = 2LL * V3 * 128 * (T + 1);
        pad_time_kernel<<<(int)((tot + 255) / 256), 256, 0, stream>>>(G0, G1, T, tot);
        hipMemsetAsync(G2, 0, (size_t)tot * 4, stream);
    }
    scatter(stream, G1, ei3, ea3, G2, V3, 128, T + 1, E3);
    chan(stream, G2, G0, V3, 128, 64, Wdc4, bdc4, 1, T + 1, 1);

    // up2: P32, cin=64, ei2, dc3 (64->32)
    pool(stream, G0, G1, P32, V2, V3, 64 * T);
    {
        long long tot = 2LL * V2 * 64 * (T + 1);
        pad_time_kernel<<<(int)((tot + 255) / 256), 256, 0, stream>>>(G1, G2, T, tot);
        hipMemsetAsync(G0, 0, (size_t)tot * 4, stream);
    }
    scatter(stream, G2, ei2, ea2, G0, V2, 64, T + 1, E2);
    chan(stream, G0, G1, V2, 64, 32, Wdc3, bdc3, 1, T + 1, 1);

    // up3: P21, cin=32, ei1, dc2 (32->16)
    pool(stream, G1, G2, P21, V1, V2, 32 * T);
    {
        long long tot = 2LL * V1 * 32 * (T + 1);
        pad_time_kernel<<<(int)((tot + 255) / 256), 256, 0, stream>>>(G2, G0, T, tot);
        hipMemsetAsync(G1, 0, (size_t)tot * 4, stream);
    }
    scatter(stream, G0, ei1, ea1, G1, V1, 32, T + 1, E1);
    chan(stream, G1, G2, V1, 32, 16, Wdc2, bdc2, 1, T + 1, 1);

    // up4: P10, cin=16, ei0, dc1 (16->1) -> d_out [2,6000,150]
    pool(stream, G2, G0, P10, V0, V1, 16 * T);
    {
        long long tot = 2LL * V0 * 16 * (T + 1);
        pad_time_kernel<<<(int)((tot + 255) / 256), 256, 0, stream>>>(G0, G1, T, tot);
        hipMemsetAsync(G2, 0, (size_t)tot * 4, stream);
    }
    scatter(stream, G1, ei0, ea0, G2, V0, 16, T + 1, E0);
    chan(stream, G2, (float*)d_out, V0, 16, 1, Wdc1, bdc1, 1, T + 1, 1);
}
